// GraphQuadratureNet_19267223289970
// MI455X (gfx1250) — matrix operand order repaired and checked by reading, not compiled, not run on hardware
//
#include <hip/hip_runtime.h>
#include <hip/hip_bf16.h>
#include <math.h>

// ---------------------------------------------------------------------------
// GraphQuadratureNet on gfx1250 (MI455X, wave32, WMMA)
//   3x [ GEMM(f16 WMMA, f32 acc) -> edge scatter-add (f32 atomics, L2-resident)
//        -> self-loop + bias + ReLU ] -> head projections + softplus
// ---------------------------------------------------------------------------

typedef __attribute__((ext_vector_type(16))) _Float16 v16h;
typedef __attribute__((ext_vector_type(8)))  float    v8f;

// ---------------- utility kernels ----------------

__global__ void k_zero_f32(float* __restrict__ p, long long n) {
    long long t = (long long)blockIdx.x * blockDim.x + threadIdx.x;
    if (t < n) p[t] = 0.0f;
}

__global__ void k_zero_i32(int* __restrict__ p, int n) {
    int t = blockIdx.x * blockDim.x + threadIdx.x;
    if (t < n) p[t] = 0;
}

__global__ void k_cvt_f32_to_f16(const float* __restrict__ in,
                                 _Float16* __restrict__ out, long long n) {
    long long t = (long long)blockIdx.x * blockDim.x + threadIdx.x;
    if (t < n) out[t] = (_Float16)in[t];
}

// Pack W [K x 64] (row-major f32) into this file's WMMA B-operand lane layout (contiguous K, matching the A loads):
// index = (((kstep*4 + nt)*32 + lane)*16 + e)
//   K   = kstep*32 + (lane>>4)*16 + e   (halves within a lane are contiguous K)
//   col = nt*16 + (lane&15)
__global__ void k_pack_w(const float* __restrict__ W, _Float16* __restrict__ out, int K) {
    int t = blockIdx.x * blockDim.x + threadIdx.x;
    int total = K * 64;
    if (t >= total) return;
    int e     = t & 15;
    int lane  = (t >> 4) & 31;
    int nt    = (t >> 9) & 3;
    int kstep = t >> 11;
    int k   = kstep * 32 + (lane >> 4) * 16 + e;
    int col = nt * 16 + (lane & 15);
    out[t] = (_Float16)W[k * 64 + col];
}

// ---------------- degree / norm ----------------

__global__ void k_deg_count(const int* __restrict__ dst, int* __restrict__ deg, int E) {
    int t = blockIdx.x * blockDim.x + threadIdx.x;
    if (t < E) atomicAdd(&deg[dst[t]], 1);
}

__global__ void k_dinv(const int* __restrict__ deg, float* __restrict__ dinv, int n) {
    int t = blockIdx.x * blockDim.x + threadIdx.x;
    if (t < n) dinv[t] = rsqrtf((float)deg[t] + 1.0f);  // +1 self loop
}

// ---------------- WMMA GEMM: C[n x 64] = A[n x K] @ Wpack ----------------
// Block = 256 threads = 8 waves; each wave computes a 16x64 output tile
// (4 accumulators of 16x16), K stepped by 32 with v_wmma_f32_16x16x32_f16.

struct AHalves { uint4 lo, hi; };

__global__ __launch_bounds__(256)
void k_gemm_wmma(const _Float16* __restrict__ A, const _Float16* __restrict__ Bp,
                 float* __restrict__ C, int n, int K) {
    const int lane = threadIdx.x & 31;
    const int wave = threadIdx.x >> 5;
    const int row0 = blockIdx.x * 128 + wave * 16;
    const int m    = lane & 15;
    const int hi   = lane >> 4;

    int arow = row0 + m;
    if (arow >= n) arow = n - 1;               // clamp (keeps EXEC uniform)
    const _Float16* __restrict__ arp = A + (size_t)arow * K;

    v8f acc0 = {}, acc1 = {}, acc2 = {}, acc3 = {};

    const int ksteps = K >> 5;
    for (int ks = 0; ks < ksteps; ++ks) {
        const int kk = ks << 5;
        // A operand: halves 0..15 <- K = kk + hi*16 .. +15 (contiguous, matches packed B)
        AHalves ab;
        ab.lo = *reinterpret_cast<const uint4*>(arp + kk + hi * 16);
        ab.hi = *reinterpret_cast<const uint4*>(arp + kk + hi * 16 + 8);
        v16h a = __builtin_bit_cast(v16h, ab);

        // B operands: pre-packed, one contiguous 32B load per lane per N-tile
        const _Float16* __restrict__ bbase = Bp + ((size_t)ks * 4 * 32 + lane) * 16;
        v16h b0 = *reinterpret_cast<const v16h*>(bbase + 0 * 32 * 16);
        v16h b1 = *reinterpret_cast<const v16h*>(bbase + 1 * 32 * 16);
        v16h b2 = *reinterpret_cast<const v16h*>(bbase + 2 * 32 * 16);
        v16h b3 = *reinterpret_cast<const v16h*>(bbase + 3 * 32 * 16);

        acc0 = __builtin_amdgcn_wmma_f32_16x16x32_f16(false, a, false, b0, (short)0, acc0, false, false);
        acc1 = __builtin_amdgcn_wmma_f32_16x16x32_f16(false, a, false, b1, (short)0, acc1, false, false);
        acc2 = __builtin_amdgcn_wmma_f32_16x16x32_f16(false, a, false, b2, (short)0, acc2, false, false);
        acc3 = __builtin_amdgcn_wmma_f32_16x16x32_f16(false, a, false, b3, (short)0, acc3, false, false);
    }

    // C/D layout: VGPR r -> row = row0 + r + hi*8, col = nt*16 + m
#pragma unroll
    for (int r = 0; r < 8; ++r) {
        int row = row0 + r + hi * 8;
        if (row < n) {
            float* crow = C + (size_t)row * 64 + m;
            crow[0]  = acc0[r];
            crow[16] = acc1[r];
            crow[32] = acc2[r];
            crow[48] = acc3[r];
        }
    }
}

// ---------------- edge aggregation: agg[dst] += xw[src] * dinv[s]*dinv[d] ----------------
// 16 threads per edge, float4 per thread -> coalesced 64-ch rows; atomics land in L2.

__global__ void k_agg_edges(const float* __restrict__ xw, const int* __restrict__ src,
                            const int* __restrict__ dst, const float* __restrict__ dinv,
                            float* __restrict__ agg, int E) {
    long long t = (long long)blockIdx.x * blockDim.x + threadIdx.x;
    int e = (int)(t >> 4);
    if (e >= E) return;
    int c = ((int)t & 15) * 4;
    int s = src[e], d = dst[e];
    float nrm = dinv[s] * dinv[d];
    const float4 v = *reinterpret_cast<const float4*>(xw + (size_t)s * 64 + c);
    float* o = agg + (size_t)d * 64 + c;
    atomicAdd(o + 0, v.x * nrm);
    atomicAdd(o + 1, v.y * nrm);
    atomicAdd(o + 2, v.z * nrm);
    atomicAdd(o + 3, v.w * nrm);
}

// self-loop contribution + bias + ReLU; emit f32 (for heads) and f16 (next GEMM A)
__global__ void k_finalize(const float* __restrict__ agg, const float* __restrict__ xw,
                           const float* __restrict__ dinv, const float* __restrict__ bias,
                           float* __restrict__ h, _Float16* __restrict__ hf, long long total) {
    long long t = (long long)blockIdx.x * blockDim.x + threadIdx.x;
    if (t >= total) return;
    int i = (int)(t >> 6);
    int c = (int)(t & 63);
    float di = dinv[i];
    float v = agg[t] + xw[t] * di * di + bias[c];
    v = fmaxf(v, 0.0f);
    h[t]  = v;
    hf[t] = (_Float16)v;
}

// ---------------- heads: shifts [n,2] then softplus weights [n] ----------------

__global__ void k_heads(const float* __restrict__ h, const float* __restrict__ Ws,
                        const float* __restrict__ bs, const float* __restrict__ Ww,
                        const float* __restrict__ bw, float* __restrict__ out, int n) {
    int i = blockIdx.x * blockDim.x + threadIdx.x;
    if (i >= n) return;
    const float* hr = h + (size_t)i * 64;
    float s0 = 0.f, s1 = 0.f, w = 0.f;
#pragma unroll
    for (int c = 0; c < 64; ++c) {
        float hv = hr[c];
        s0 += hv * Ws[c * 2 + 0];
        s1 += hv * Ws[c * 2 + 1];
        w  += hv * Ww[c];
    }
    out[(size_t)i * 2 + 0] = s0 + bs[0];
    out[(size_t)i * 2 + 1] = s1 + bs[1];
    float z = w + bw[0];
    float sp = (z > 20.0f) ? z : log1pf(__expf(z));   // stable softplus
    out[(size_t)n * 2 + i] = sp;
}

// ---------------------------------------------------------------------------

static inline size_t align_up(size_t v, size_t a) { return (v + a - 1) & ~(a - 1); }
static inline int    cdiv(long long a, long long b) { return (int)((a + b - 1) / b); }

extern "C" void kernel_launch(void* const* d_in, const int* in_sizes, int n_in,
                              void* d_out, int out_size, void* d_ws, size_t ws_size,
                              hipStream_t stream) {
    const float* x    = (const float*)d_in[0];
    const int*   eidx = (const int*)d_in[1];
    const float* W1 = (const float*)d_in[2];  const float* b1 = (const float*)d_in[3];
    const float* W2 = (const float*)d_in[4];  const float* b2 = (const float*)d_in[5];
    const float* W3 = (const float*)d_in[6];  const float* b3 = (const float*)d_in[7];
    const float* Ws = (const float*)d_in[8];  const float* bs = (const float*)d_in[9];
    const float* Ww = (const float*)d_in[10]; const float* bw = (const float*)d_in[11];

    const int n = in_sizes[0] / 128;   // 50000
    const int E = in_sizes[1] / 2;     // 800000
    const int*   src = eidx;
    const int*   dst = eidx + E;

    // ---- carve workspace ----
    char* ws = (char*)d_ws;
    size_t off = 0;
    auto carve = [&](size_t bytes) -> void* {
        void* p = ws + off;
        off = align_up(off + bytes, 256);
        return p;
    };
    int*      deg  = (int*)carve((size_t)n * 4);
    float*    dinv = (float*)carve((size_t)n * 4);
    _Float16* xh   = (_Float16*)carve((size_t)n * 128 * 2);
    _Float16* hf   = (_Float16*)carve((size_t)n * 64 * 2);
    float*    xw   = (float*)carve((size_t)n * 64 * 4);
    float*    agg  = (float*)carve((size_t)n * 64 * 4);
    float*    hbuf = (float*)carve((size_t)n * 64 * 4);
    _Float16* w1p  = (_Float16*)carve(128 * 64 * 2);
    _Float16* w2p  = (_Float16*)carve(64 * 64 * 2);
    _Float16* w3p  = (_Float16*)carve(64 * 64 * 2);
    (void)ws_size; (void)n_in; (void)out_size;

    const int TB = 256;
    const long long nh = (long long)n * 64;

    // ---- degrees + symmetric norm factors ----
    k_zero_i32<<<cdiv(n, TB), TB, 0, stream>>>(deg, n);
    k_deg_count<<<cdiv(E, TB), TB, 0, stream>>>(dst, deg, E);
    k_dinv<<<cdiv(n, TB), TB, 0, stream>>>(deg, dinv, n);

    // ---- precision conversion / weight packing ----
    k_cvt_f32_to_f16<<<cdiv((long long)n * 128, TB), TB, 0, stream>>>(x, xh, (long long)n * 128);
    k_pack_w<<<cdiv(128 * 64, TB), TB, 0, stream>>>(W1, w1p, 128);
    k_pack_w<<<cdiv(64 * 64, TB), TB, 0, stream>>>(W2, w2p, 64);
    k_pack_w<<<cdiv(64 * 64, TB), TB, 0, stream>>>(W3, w3p, 64);

    const int gemm_blocks = cdiv(n, 128);
    const int agg_blocks  = cdiv((long long)E * 16, TB);

    // ---- layer 1 ----
    k_gemm_wmma<<<gemm_blocks, TB, 0, stream>>>(xh, w1p, xw, n, 128);
    k_zero_f32<<<cdiv(nh, TB), TB, 0, stream>>>(agg, nh);
    k_agg_edges<<<agg_blocks, TB, 0, stream>>>(xw, src, dst, dinv, agg, E);
    k_finalize<<<cdiv(nh, TB), TB, 0, stream>>>(agg, xw, dinv, b1, hbuf, hf, nh);

    // ---- layer 2 ----
    k_gemm_wmma<<<gemm_blocks, TB, 0, stream>>>(hf, w2p, xw, n, 64);
    k_zero_f32<<<cdiv(nh, TB), TB, 0, stream>>>(agg, nh);
    k_agg_edges<<<agg_blocks, TB, 0, stream>>>(xw, src, dst, dinv, agg, E);
    k_finalize<<<cdiv(nh, TB), TB, 0, stream>>>(agg, xw, dinv, b2, hbuf, hf, nh);

    // ---- layer 3 ----
    k_gemm_wmma<<<gemm_blocks, TB, 0, stream>>>(hf, w3p, xw, n, 64);
    k_zero_f32<<<cdiv(nh, TB), TB, 0, stream>>>(agg, nh);
    k_agg_edges<<<agg_blocks, TB, 0, stream>>>(xw, src, dst, dinv, agg, E);
    k_finalize<<<cdiv(nh, TB), TB, 0, stream>>>(agg, xw, dinv, b3, hbuf, hf, nh);

    // ---- heads ----
    k_heads<<<cdiv(n, TB), TB, 0, stream>>>(hbuf, Ws, bs, Ww, bw, (float*)d_out, n);
}
